// SimiNet_20976620273819
// MI455X (gfx1250) — matrix operand order repaired and checked by reading, not compiled, not run on hardware
//
#include <hip/hip_runtime.h>
#include <hip/hip_bf16.h>

typedef __attribute__((ext_vector_type(16))) _Float16     v16h;
typedef __attribute__((ext_vector_type(8)))  float        v8f;
typedef __attribute__((ext_vector_type(4)))  unsigned int v4u;
typedef __attribute__((ext_vector_type(4)))  int          v4i;
typedef __attribute__((ext_vector_type(8)))  int          v8i;

#if defined(__has_builtin)
# if __has_builtin(__builtin_amdgcn_tensor_load_to_lds)
#  define HAVE_TDM 1
# else
#  define HAVE_TDM 0
# endif
# if __has_builtin(__builtin_amdgcn_s_wait_tensorcnt)
#  define HAVE_WTC 1
# else
#  define HAVE_WTC 0
# endif
#else
# define HAVE_TDM 0
# define HAVE_WTC 0
#endif

// ---------------------------------------------------------------------------
// WMMA implicit-GEMM conv.  Workgroup = 256 threads (8 waves), computes a
// 16(Cout) x 256(pixel) tile; each wave owns 16x32 pixels via two f32
// accumulators so every A (weight) fragment feeds two v_wmma ops.
//
// The 16-row weight block  wgt[cobase..cobase+15][*]  (16 contiguous rows of
// N0 = Cin*Ksz*Ksz floats) is staged into LDS once per workgroup with the
// Tensor Data Mover (TENSOR_LOAD_TO_LDS, D# per ISA ch.8: 2D tile, 16 rows,
// row length/stride N0, data_size=4B), completion via s_wait_tensorcnt then a
// workgroup barrier.  Falls back to a cooperative load when the builtin is
// unavailable.
//
// Fragment layouts per ISA 7.12.2 (16-bit operands):
//   A 16x32 MxK:  lane&15 = M;  halves i: K = (lane>=16?16:0) + i  (used here, not ISA table order)
//   B 32x16 KxN:  lane&15 = N;  halves i: K = (lane>=16?16:0) + i  (same as A; not ISA table order)
//   C/D 16x16 f32: vgpr v, lane: N = lane&15, M = v + (lane>=16?8:0)
// ---------------------------------------------------------------------------
__global__ void __launch_bounds__(256)
conv_wmma_k(const float* __restrict__ in, const float* __restrict__ wgt,
            const float* __restrict__ bias, float* __restrict__ out,
            int Cin, int Cout, int H, int W, int Ksz, int pad, int relu)
{
    extern __shared__ float smem[];          // [16][N0] f32 weight block
    const int tid  = threadIdx.x;
    const int lane = tid & 31;
    const int wid  = tid >> 5;               // wave 0..7
    const int grp  = lane >> 4;
    const int l    = lane & 15;
    const int HW   = H * W;
    const int KK   = Ksz * Ksz;
    const int N0   = Cin * KK;
    const int cobase = blockIdx.y * 16;
    const int b      = blockIdx.z;
    const float* wrow = wgt + (size_t)cobase * N0;   // 16 contiguous rows

#if HAVE_TDM
    if (tid == 0) {
        uint64_t ga = (uint64_t)(const void*)wrow;   // 57-bit global byte addr
        v4u g0;
        g0[0] = 1u;                                   // count=1, no gather
        g0[1] = (unsigned)(size_t)smem;               // lds_addr (bytes)
        g0[2] = (unsigned)(ga & 0xffffffffu);         // global_addr[95:64]
        g0[3] = ((unsigned)(ga >> 32) & 0x01ffffffu)  // global_addr[120:96]
                | (2u << 30);                         // type=2 ("image")
        v8i g1;
        g1[0] = 0x00020000;                           // data_size=2 -> 4 bytes
        g1[1] = (int)(((unsigned)N0 & 0xffffu) << 16);          // tensor_dim0 lo
        g1[2] = (int)(((unsigned)N0 >> 16) |
                      (((unsigned)Cout & 0xffffu) << 16));      // dim0 hi | tensor_dim1 lo
        g1[3] = (int)(((unsigned)N0 & 0xffffu) << 16);          // dim1 hi(0) | tile_dim0=N0
        g1[4] = 16;                                   // tile_dim1=16, tile_dim2=0
        g1[5] = N0;                                   // tensor_dim0_stride[31:0]
        g1[6] = 0;                                    // stride0 hi | stride1 lo
        g1[7] = 0;                                    // stride1 hi
        v4i gz = {0, 0, 0, 0};
#if __clang_major__ >= 23
        v8i gz8 = {0, 0, 0, 0, 0, 0, 0, 0};
        __builtin_amdgcn_tensor_load_to_lds(g0, g1, gz, gz, gz8, 0);
#else
        __builtin_amdgcn_tensor_load_to_lds(g0, g1, gz, gz, 0);
#endif
    }
#if HAVE_WTC
    __builtin_amdgcn_s_wait_tensorcnt(0);
#else
    asm volatile("s_wait_tensorcnt 0x0" ::: "memory");
#endif
#else
    for (int i = tid; i < 16 * N0; i += 256) smem[i] = wrow[i];
#endif
    __syncthreads();

    const int p0  = blockIdx.x * 256 + wid * 32 + l;  // this lane's two N columns
    const int p1  = p0 + 16;
    const int oh0 = p0 / W, ow0 = p0 % W;
    const int oh1 = p1 / W, ow1 = p1 % W;

    v8f acc0 = {}, acc1 = {};
    for (int r = 0; r < Ksz; ++r) {
        for (int s = 0; s < Ksz; ++s) {
            const int y0 = oh0 + r - pad, x0 = ow0 + s - pad;
            const int y1 = oh1 + r - pad, x1 = ow1 + s - pad;
            const bool ib0 = ((unsigned)y0 < (unsigned)H) && ((unsigned)x0 < (unsigned)W);
            const bool ib1 = ((unsigned)y1 < (unsigned)H) && ((unsigned)x1 < (unsigned)W);
            const float* ip0 = in + (size_t)b * Cin * HW + (long long)y0 * W + x0;
            const float* ip1 = in + (size_t)b * Cin * HW + (long long)y1 * W + x1;
            const float* wl  = smem + (size_t)l * N0 + r * Ksz + s;   // LDS row of M=l
            for (int cb = 0; cb < Cin; cb += 32) {
                v16h A, B0, B1;
#pragma unroll
                for (int i = 0; i < 16; ++i) {
                    int koff = grp * 16 + i;
                    int ci = cb + koff;
                    float av = (ci < Cin) ? wl[(size_t)ci * KK] : 0.0f;  // ds_load
                    A[i] = (_Float16)av;
                }
#pragma unroll
                for (int i = 0; i < 16; ++i) {
                    int ci = cb + grp * 16 + i;
                    bool ok = (ci < Cin);
                    float b0v = (ib0 && ok) ? ip0[(size_t)ci * HW] : 0.0f;
                    float b1v = (ib1 && ok) ? ip1[(size_t)ci * HW] : 0.0f;
                    B0[i] = (_Float16)b0v;
                    B1[i] = (_Float16)b1v;
                }
                acc0 = __builtin_amdgcn_wmma_f32_16x16x32_f16(
                           false, A, false, B0, (short)0, acc0, false, false);
                acc1 = __builtin_amdgcn_wmma_f32_16x16x32_f16(
                           false, A, false, B1, (short)0, acc1, false, false);
            }
        }
    }

#pragma unroll
    for (int v = 0; v < 8; ++v) {
        int co = cobase + grp * 8 + v;
        float bv = bias[co];
        float v0 = acc0[v] + bv;
        float v1 = acc1[v] + bv;
        if (relu) { v0 = fmaxf(v0, 0.0f); v1 = fmaxf(v1, 0.0f); }
        out[((size_t)b * Cout + co) * HW + p0] = v0;
        out[((size_t)b * Cout + co) * HW + p1] = v1;
    }
}

// ---------------------------------------------------------------------------
// Self-correlation GEMM: per batch, X [256 x 512] (raw reshape of NCHW
// [512,16,16] memory), corr = X * X^T / 512  -> [256 x 256].  WMMA f16.
// ---------------------------------------------------------------------------
__global__ void __launch_bounds__(32)
corr_wmma_k(const float* __restrict__ X, float* __restrict__ corr)
{
    const int lane = threadIdx.x, grp = lane >> 4, l = lane & 15;
    const int nbase = blockIdx.x * 16;
    const int mbase = blockIdx.y * 16;
    const int b     = blockIdx.z;
    const float* Xb = X + (size_t)b * 256 * 512;

    v8f acc = {};
    for (int cb = 0; cb < 512; cb += 32) {
        v16h A, Bf;
#pragma unroll
        for (int i = 0; i < 16; ++i) {
            int koff = grp * 16 + i;
            A[i] = (_Float16)Xb[(size_t)(mbase + l) * 512 + cb + koff];
        }
#pragma unroll
        for (int i = 0; i < 16; ++i) {
            Bf[i] = (_Float16)Xb[(size_t)(nbase + l) * 512 + cb + grp * 16 + i];
        }
        acc = __builtin_amdgcn_wmma_f32_16x16x32_f16(
                  false, A, false, Bf, (short)0, acc, false, false);
    }
#pragma unroll
    for (int v = 0; v < 8; ++v) {
        int m = mbase + grp * 8 + v;
        corr[((size_t)b * 256 + m) * 256 + nbase + l] = acc[v] * (1.0f / 512.0f);
    }
}

// ---------------------------------------------------------------------------
// Per-row descending bitonic sort of corr[b,m,:] (256) + percentile-rank
// gather, stored transposed: out[b, r, h, w] with m = h*16+w.
// ---------------------------------------------------------------------------
__global__ void __launch_bounds__(256)
sortpool_k(const float* __restrict__ corr, float* __restrict__ out)
{
    __shared__ float s[256];
    const int t   = threadIdx.x;
    const int row = blockIdx.x;          // b*256 + m
    const int b   = row >> 8;
    const int m   = row & 255;

    s[t] = corr[(size_t)row * 256 + t];
    for (int k = 2; k <= 256; k <<= 1) {
        for (int j = k >> 1; j > 0; j >>= 1) {
            __syncthreads();
            int ixj = t ^ j;
            if (ixj > t) {
                float a = s[t], c = s[ixj];
                bool desc = ((t & k) == 0);
                if (desc ? (a < c) : (a > c)) { s[t] = c; s[ixj] = a; }
            }
        }
    }
    __syncthreads();
    int rk = (int)rintf(1.0f + (float)t * (254.0f / 255.0f));
    out[((size_t)b << 16) + ((size_t)t << 8) + m] = s[rk];
}

// ---------------------------------------------------------------------------
// LayerNorm over the last 16x16 per (b,c) row; w/b are [16,16] flat.
// ---------------------------------------------------------------------------
__global__ void __launch_bounds__(256)
ln16_k(const float* __restrict__ x, const float* __restrict__ w,
       const float* __restrict__ bb, float* __restrict__ y)
{
    __shared__ float s1[256], s2[256];
    const int t = threadIdx.x;
    const size_t base = (size_t)blockIdx.x * 256;
    float v = x[base + t];
    s1[t] = v; s2[t] = v * v;
    for (int off = 128; off > 0; off >>= 1) {
        __syncthreads();
        if (t < off) { s1[t] += s1[t + off]; s2[t] += s2[t + off]; }
    }
    __syncthreads();
    float mean = s1[0] * (1.0f / 256.0f);
    float var  = s2[0] * (1.0f / 256.0f) - mean * mean;
    float inv  = rsqrtf(var + 1e-5f);
    y[base + t] = (v - mean) * inv * w[t] + bb[t];
}

// ---------------------------------------------------------------------------
// Training-mode BatchNorm over (B,H,W) per channel, optional ReLU, output
// written at channel offset inside a (possibly larger) concat tensor.
// ---------------------------------------------------------------------------
__global__ void __launch_bounds__(256)
bn_relu_k(const float* __restrict__ x, float* __restrict__ y,
          const float* __restrict__ g, const float* __restrict__ bt,
          int B, int C, int HW, int outCtot, int outOff, int relu)
{
    __shared__ float s1[256], s2[256];
    const int t = threadIdx.x, c = blockIdx.x;
    const long long N = (long long)B * HW;
    float a = 0.0f, q = 0.0f;
    for (long long i = t; i < N; i += 256) {
        int bb = (int)(i / HW), hw = (int)(i % HW);
        float v = x[((size_t)bb * C + c) * HW + hw];
        a += v; q += v * v;
    }
    s1[t] = a; s2[t] = q;
    for (int off = 128; off > 0; off >>= 1) {
        __syncthreads();
        if (t < off) { s1[t] += s1[t + off]; s2[t] += s2[t + off]; }
    }
    __syncthreads();
    float mean  = s1[0] / (float)N;
    float var   = s2[0] / (float)N - mean * mean;
    float scale = g[c] * rsqrtf(var + 1e-5f);
    float shift = bt[c] - mean * scale;
    for (long long i = t; i < N; i += 256) {
        int bb = (int)(i / HW), hw = (int)(i % HW);
        float v = x[((size_t)bb * C + c) * HW + hw] * scale + shift;
        if (relu) v = fmaxf(v, 0.0f);
        y[((size_t)bb * outCtot + outOff + c) * HW + hw] = v;
    }
}

// ---------------------------------------------------------------------------
// Direct fp32 conv for small-channel decoder layers; writes into a concat
// tensor at channel offset outOff (channel total outCtot).
// ---------------------------------------------------------------------------
__global__ void __launch_bounds__(256)
conv_direct_k(const float* __restrict__ in, const float* __restrict__ wgt,
              const float* __restrict__ bias, float* __restrict__ out,
              int B, int Cin, int H, int W, int Cout, int Ksz, int pad,
              int outCtot, int outOff, int relu)
{
    long long total = (long long)B * Cout * H * W;
    long long idx = (long long)blockIdx.x * 256 + threadIdx.x;
    if (idx >= total) return;
    int x  = (int)(idx % W);
    int h  = (int)((idx / W) % H);
    int co = (int)((idx / ((long long)W * H)) % Cout);
    int b  = (int)(idx / ((long long)W * H * Cout));

    float acc = bias[co];
    for (int ci = 0; ci < Cin; ++ci) {
        const float* ip = in + ((size_t)b * Cin + ci) * H * W;
        const float* wp = wgt + ((size_t)co * Cin + ci) * Ksz * Ksz;
        for (int r = 0; r < Ksz; ++r) {
            int yy = h + r - pad;
            if ((unsigned)yy >= (unsigned)H) continue;
            for (int s = 0; s < Ksz; ++s) {
                int xx = x + s - pad;
                if ((unsigned)xx >= (unsigned)W) continue;
                acc += ip[(size_t)yy * W + xx] * wp[r * Ksz + s];
            }
        }
    }
    if (relu) acc = fmaxf(acc, 0.0f);
    out[((size_t)b * outCtot + outOff + co) * H * W + (size_t)h * W + x] = acc;
}

// ---------------------------------------------------------------------------
// Bilinear x2 upsample, align_corners=True, with input/output channel slices.
// ---------------------------------------------------------------------------
__global__ void __launch_bounds__(256)
up2_k(const float* __restrict__ in, float* __restrict__ out,
      int B, int C, int H, int W, int inCtot, int inOff, int outCtot, int outOff)
{
    const int OH = 2 * H, OW = 2 * W;
    long long total = (long long)B * C * OH * OW;
    long long idx = (long long)blockIdx.x * 256 + threadIdx.x;
    if (idx >= total) return;
    int ox = (int)(idx % OW);
    int oy = (int)((idx / OW) % OH);
    int c  = (int)((idx / ((long long)OW * OH)) % C);
    int b  = (int)(idx / ((long long)OW * OH * C));

    float sy = (float)oy * (float)(H - 1) / (float)(OH - 1);
    float sx = (float)ox * (float)(W - 1) / (float)(OW - 1);
    int y0 = (int)floorf(sy); int y1 = (y0 + 1 < H) ? y0 + 1 : H - 1; float fy = sy - (float)y0;
    int x0 = (int)floorf(sx); int x1 = (x0 + 1 < W) ? x0 + 1 : W - 1; float fx = sx - (float)x0;

    const float* ip = in + ((size_t)b * inCtot + inOff + c) * H * W;
    float v00 = ip[(size_t)y0 * W + x0], v01 = ip[(size_t)y0 * W + x1];
    float v10 = ip[(size_t)y1 * W + x0], v11 = ip[(size_t)y1 * W + x1];
    float r0 = v00 * (1.0f - fx) + v01 * fx;
    float r1 = v10 * (1.0f - fx) + v11 * fx;
    out[((size_t)b * outCtot + outOff + c) * OH * OW + (size_t)oy * OW + ox] =
        r0 * (1.0f - fy) + r1 * fy;
}

// ---------------------------------------------------------------------------
// 2x2 max pool, stride 2.
// ---------------------------------------------------------------------------
__global__ void __launch_bounds__(256)
maxpool2_k(const float* __restrict__ in, float* __restrict__ out,
           int B, int C, int H, int W)
{
    const int OH = H / 2, OW = W / 2;
    long long total = (long long)B * C * OH * OW;
    long long idx = (long long)blockIdx.x * 256 + threadIdx.x;
    if (idx >= total) return;
    int ox = (int)(idx % OW);
    int oy = (int)((idx / OW) % OH);
    int c  = (int)((idx / ((long long)OW * OH)) % C);
    int b  = (int)(idx / ((long long)OW * OH * C));
    const float* ip = in + ((size_t)b * C + c) * H * W;
    int y = oy * 2, x = ox * 2;
    float m = fmaxf(fmaxf(ip[(size_t)y * W + x], ip[(size_t)y * W + x + 1]),
                    fmaxf(ip[(size_t)(y + 1) * W + x], ip[(size_t)(y + 1) * W + x + 1]));
    out[((size_t)b * C + c) * OH * OW + (size_t)oy * OW + ox] = m;
}

// ===========================================================================
extern "C" void kernel_launch(void* const* d_in, const int* in_sizes, int n_in,
                              void* d_out, int out_size, void* d_ws, size_t ws_size,
                              hipStream_t stream)
{
    // Allow the 512-channel weight blocks (16*512*9*4 B) as dynamic LDS.
    (void)hipFuncSetAttribute(reinterpret_cast<const void*>(conv_wmma_k),
                              hipFuncAttributeMaxDynamicSharedMemorySize,
                              16 * 512 * 9 * 4);

    auto F = [&](int i) -> const float* { return (const float*)d_in[i]; };

    // ---- input mapping: insertion-order vs sorted-pytree flattening -------
    const bool insertion = (n_in > 0 && in_sizes[0] == 4 * 3 * 256 * 256);
    const float *x, *mask_w, *mask_b, *ln_w_, *ln_b_, *bn1_g, *bn1_b;
    const float *cw[10], *cb_[10]; // c1a,c1b,c2a,c2b,c3a,c3b,c3c,c4a,c4b,c4c
    struct BiP { const float *w[3], *b[3], *g, *bt; } bi[6];

    if (insertion) {
        x = F(0);
        for (int i = 0; i < 10; ++i) { cw[i] = F(1 + 2 * i); cb_[i] = F(2 + 2 * i); }
        mask_w = F(21); mask_b = F(22);
        ln_w_  = F(23); ln_b_  = F(24);
        bn1_g  = F(25); bn1_b  = F(26);
        for (int k = 0; k < 6; ++k) {
            int b0 = 27 + 8 * k;
            bi[k].w[0] = F(b0 + 0); bi[k].b[0] = F(b0 + 1);
            bi[k].w[1] = F(b0 + 2); bi[k].b[1] = F(b0 + 3);
            bi[k].w[2] = F(b0 + 4); bi[k].b[2] = F(b0 + 5);
            bi[k].g    = F(b0 + 6); bi[k].bt   = F(b0 + 7);
        }
    } else { // sorted keys: bi1..bi6{b0,b1,b2,bt,g,w0,w1,w2}, bn1_b, bn1_g, c*_b/w, ln, mask, x
        for (int k = 0; k < 6; ++k) {
            int b0 = 8 * k;
            bi[k].b[0] = F(b0 + 0); bi[k].b[1] = F(b0 + 1); bi[k].b[2] = F(b0 + 2);
            bi[k].bt   = F(b0 + 3); bi[k].g    = F(b0 + 4);
            bi[k].w[0] = F(b0 + 5); bi[k].w[1] = F(b0 + 6); bi[k].w[2] = F(b0 + 7);
        }
        bn1_b = F(48); bn1_g = F(49);
        for (int i = 0; i < 10; ++i) { cb_[i] = F(50 + 2 * i); cw[i] = F(51 + 2 * i); }
        ln_b_ = F(70); ln_w_ = F(71);
        mask_b = F(72); mask_w = F(73);
        x = F(74);
    }

    // ---- workspace layout (floats) ----------------------------------------
    float* Wb = (float*)d_ws;
    size_t off = 0;
    auto alloc = [&](size_t n) { size_t o = off; off += (n + 63) & ~(size_t)63; return o; };
    const size_t o_e0  = alloc(16777216);  // encoder ping
    const size_t o_e1  = alloc(16777216);  // encoder pong
    const size_t o_ln  = alloc(524288);    // layernorm'd h  [4,512,16,16]
    const size_t o_cor = alloc(262144);    // corr           [4,256,256]
    const size_t o_pl  = alloc(262144);    // pooled         [4,256,16,16]
    const size_t o_bn  = alloc(262144);    // bn1 output
    const size_t o_d1r = alloc(24576),  o_d1  = alloc(24576);    // f16  [4,24,16,16]
    const size_t o_d2  = alloc(98304);                            // f32  [4,24,32,32]
    const size_t o_d3r = alloc(73728),  o_d3  = alloc(73728);    // dx32 [4,18,32,32]
    const size_t o_d4  = alloc(688128);                           // cat64 [4,42,64,64]
    const size_t o_d5r = alloc(196608), o_d5  = alloc(196608);   // dx64 [4,12,64,64]
    const size_t o_d6  = alloc(2359296);                          // cat128 [4,36,128,128]
    const size_t o_d7r = alloc(393216), o_d7  = alloc(393216);   // dx128 [4,6,128,128]
    const size_t o_d8  = alloc(7864320);                          // cat256 [4,30,256,256]
    const size_t o_d10 = alloc(7864320);                          // catmask [4,30,256,256]
    const size_t o_d9r = alloc(1572864);                          // bi5 raw [4,6,256,256]
    const size_t o_11r = alloc(1572864), o_11 = alloc(1572864);  // masks [4,6,256,256]
    (void)ws_size; (void)out_size;

    // ---- launch helpers ----------------------------------------------------
    auto convw = [&](const float* in, int li, float* out, int Cin, int Cout, int Hn) {
        dim3 g((unsigned)(Hn * Hn / 256), (unsigned)(Cout / 16), 4);
        size_t shm = (size_t)16 * Cin * 9 * sizeof(float);
        conv_wmma_k<<<g, dim3(256), shm, stream>>>(in, cw[li], cb_[li], out,
                                                   Cin, Cout, Hn, Hn, 3, 1, 1);
    };
    auto mp = [&](const float* in, float* out, int C, int Hn) {
        long long tot = 4LL * C * (Hn / 2) * (Hn / 2);
        maxpool2_k<<<dim3((unsigned)((tot + 255) / 256)), dim3(256), 0, stream>>>(
            in, out, 4, C, Hn, Hn);
    };
    auto cdir = [&](const float* in, const float* w, const float* b, float* out,
                    int Cin, int Hn, int Cout, int K, int pad, int Ctot, int coff, int relu) {
        long long tot = 4LL * Cout * Hn * Hn;
        conv_direct_k<<<dim3((unsigned)((tot + 255) / 256)), dim3(256), 0, stream>>>(
            in, w, b, out, 4, Cin, Hn, Hn, Cout, K, pad, Ctot, coff, relu);
    };
    auto bnr = [&](const float* xin, float* y, const float* g, const float* bt,
                   int C, int HW, int Ctot, int coff, int relu) {
        bn_relu_k<<<dim3((unsigned)C), dim3(256), 0, stream>>>(
            xin, y, g, bt, 4, C, HW, Ctot, coff, relu);
    };
    auto up = [&](const float* in, float* out, int C, int Hn,
                  int inCtot, int inOff, int outCtot, int outOff) {
        long long tot = 4LL * C * (2 * Hn) * (2 * Hn);
        up2_k<<<dim3((unsigned)((tot + 255) / 256)), dim3(256), 0, stream>>>(
            in, out, 4, C, Hn, Hn, inCtot, inOff, outCtot, outOff);
    };
    static const int ks135[3] = {1, 3, 5},  pd012[3] = {0, 1, 2};
    static const int ks571[3] = {5, 7, 11}, pd235[3] = {2, 3, 5};
    auto bib = [&](const float* in, int Cin, int Hn, int k, int o,
                   const int* ks, const int* pads, float* raw,
                   float* outp, int outCtot, int outOff) {
        for (int j = 0; j < 3; ++j)
            cdir(in, bi[k].w[j], bi[k].b[j], raw, Cin, Hn, o, ks[j], pads[j], 3 * o, j * o, 0);
        bnr(raw, outp, bi[k].g, bi[k].bt, 3 * o, Hn * Hn, outCtot, outOff, 1);
    };

    float* e0 = Wb + o_e0;
    float* e1 = Wb + o_e1;

    // ---- encoder (WMMA implicit GEMM convs + maxpools) --------------------
    convw(x,  0, e0,   3,  64, 256);
    convw(e0, 1, e1,  64,  64, 256);
    mp(e1, e0, 64, 256);                       // [4,64,128,128]
    convw(e0, 2, e1,  64, 128, 128);
    convw(e1, 3, e0, 128, 128, 128);
    mp(e0, e1, 128, 128);                      // [4,128,64,64]
    convw(e1, 4, e0, 128, 256, 64);
    convw(e0, 5, e1, 256, 256, 64);
    convw(e1, 6, e0, 256, 256, 64);
    mp(e0, e1, 256, 64);                       // [4,256,32,32]
    convw(e1, 7, e0, 256, 512, 32);
    convw(e0, 8, e1, 512, 512, 32);
    convw(e1, 9, e0, 512, 512, 32);
    mp(e0, e1, 512, 32);                       // h = [4,512,16,16] in e1

    // ---- layernorm + self-correlation pooling + bn1 ------------------------
    ln16_k<<<dim3(2048), dim3(256), 0, stream>>>(e1, ln_w_, ln_b_, Wb + o_ln);
    corr_wmma_k<<<dim3(16, 16, 4), dim3(32), 0, stream>>>(Wb + o_ln, Wb + o_cor);
    sortpool_k<<<dim3(1024), dim3(256), 0, stream>>>(Wb + o_cor, Wb + o_pl);
    bnr(Wb + o_pl, Wb + o_bn, bn1_g, bn1_b, 256, 256, 256, 0, 0);

    // ---- decoder (concats fused via channel offsets) -----------------------
    bib(Wb + o_bn, 256, 16, 0, 8, ks135, pd012, Wb + o_d1r, Wb + o_d1, 24, 0); // f16
    up(Wb + o_d1, Wb + o_d2, 24, 16, 24, 0, 24, 0);                            // f32
    bib(Wb + o_d2, 24, 32, 1, 6, ks135, pd012, Wb + o_d3r, Wb + o_d3, 18, 0);  // dx32
    up(Wb + o_d2, Wb + o_d4, 24, 32, 24, 0, 42, 0);                            // f64a
    up(Wb + o_d3, Wb + o_d4, 18, 32, 18, 0, 42, 24);                           // f64b
    bib(Wb + o_d4, 42, 64, 2, 4, ks135, pd012, Wb + o_d5r, Wb + o_d5, 12, 0);  // dx64
    up(Wb + o_d4, Wb + o_d6, 24, 64, 42, 0, 36, 0);                            // f128a
    up(Wb + o_d5, Wb + o_d6, 12, 64, 12, 0, 36, 24);                           // f128b
    bib(Wb + o_d6, 36, 128, 3, 2, ks135, pd012, Wb + o_d7r, Wb + o_d7, 6, 0);  // dx128
    up(Wb + o_d6, Wb + o_d8, 24, 128, 36, 0, 30, 0);                           // f256a -> cat256
    up(Wb + o_d7, Wb + o_d8,  6, 128,  6, 0, 30, 24);                          // f256b
    up(Wb + o_d6, Wb + o_d10, 24, 128, 36, 0, 30, 0);                          // f256a -> catmask
    for (int j = 0; j < 3; ++j)                                                // bi5 -> dx256
        cdir(Wb + o_d8, bi[4].w[j], bi[4].b[j], Wb + o_d9r,
             30, 256, 2, ks135[j], pd012[j], 6, j * 2, 0);
    bnr(Wb + o_d9r, Wb + o_d10, bi[4].g, bi[4].bt, 6, 65536, 30, 24, 1);       // into catmask
    bib(Wb + o_d10, 30, 256, 5, 2, ks571, pd235, Wb + o_11r, Wb + o_11, 6, 0); // masks
    cdir(Wb + o_11, mask_w, mask_b, (float*)d_out, 6, 256, 2, 3, 1, 2, 0, 0);  // final
}
